// MambaModel_82411832476092
// MI455X (gfx1250) — matrix operand order repaired and checked by reading, not compiled, not run on hardware
//
#include <hip/hip_runtime.h>
#include <hip/hip_bf16.h>

// ---------------------------------------------------------------------------
// Mamba model (B=4, L=512, D=768, E=1536, N=16, R=48, K=4, NL=4) for gfx1250.
// GEMMs: v_wmma_f32_16x16x32_bf16 with double-buffered LDS tiles staged via
// global_load_async_to_lds_b128 (ASYNCcnt) -> next tile prefetch overlaps WMMA.
// Selective scan: one lane per (b,e) channel, wave32 shuffle broadcast.
// ---------------------------------------------------------------------------

typedef __bf16 bf16_t;
typedef __attribute__((ext_vector_type(16))) __bf16 v16bf;
typedef __attribute__((ext_vector_type(8)))  float  v8f;

#define BB      4
#define LL      512
#define DMODEL  768
#define EE      1536
#define NN      16
#define RR      48
#define KK      4
#define NLAYER  4
#define HH      1536
#define TT      28
#define PROJW   (RR + 2 * NN)   // 80
#define ROWS    (BB * LL)       // 2048

// --- async global->LDS path (gfx1250), gated so compile never breaks -------
#if defined(__has_builtin)
#  if __has_builtin(__builtin_amdgcn_global_load_async_to_lds_b128) && \
      __has_builtin(__builtin_amdgcn_s_wait_asynccnt)
#    define USE_ASYNC_LDS 1
#  endif
#endif
#ifndef USE_ASYNC_LDS
#  define USE_ASYNC_LDS 0
#endif

// Builtin signature (probed via hipcc diagnostic): arg0 = v4i* addrspace(1),
// arg1 = v4i* addrspace(3), then two immediate ints (offset, cpol).
typedef int v4i __attribute__((vector_size(16)));
typedef __attribute__((address_space(1))) v4i* as1_v4i_p;
typedef __attribute__((address_space(3))) v4i* as3_v4i_p;

__device__ __forceinline__ float silu_f(float x) { return x / (1.f + __expf(-x)); }
__device__ __forceinline__ float softplus_f(float x) {
    return (x > 20.f) ? x : log1pf(__expf(x));
}

// ---------------------------------------------------------------------------
// Elementwise / small kernels
// ---------------------------------------------------------------------------
__global__ void k_embed(const int* __restrict__ ids, const float* __restrict__ emb,
                        float* __restrict__ h) {
    int idx = blockIdx.x * blockDim.x + threadIdx.x;
    if (idx >= ROWS * DMODEL) return;
    int d = idx % DMODEL;
    int row = idx / DMODEL;
    h[idx] = emb[(long)ids[row] * DMODEL + d];
}

__global__ void k_f32_to_bf16(const float* __restrict__ src, bf16_t* __restrict__ dst, int n) {
    int idx = blockIdx.x * blockDim.x + threadIdx.x;
    if (idx < n) dst[idx] = (bf16_t)src[idx];
}

// Convert + transpose: src is Kd x Nd row-major f32; dst is Nd x Kd row-major bf16.
__global__ void k_f32_to_bf16_T(const float* __restrict__ src, bf16_t* __restrict__ dst,
                                int Kd, int Nd) {
    int idx = blockIdx.x * blockDim.x + threadIdx.x;
    if (idx >= Kd * Nd) return;
    int k = idx % Kd;
    int n = idx / Kd;
    dst[(long)n * Kd + k] = (bf16_t)src[(long)k * Nd + n];
}

// One wave32 per row: rmsnorm(x, w) -> bf16
__global__ __launch_bounds__(256)
void k_rmsnorm_bf16(const float* __restrict__ x, const float* __restrict__ w,
                    bf16_t* __restrict__ out, int rows, int D) {
    int gw = (blockIdx.x * blockDim.x + threadIdx.x) >> 5;
    int lane = threadIdx.x & 31;
    if (gw >= rows) return;
    const float* xr = x + (long)gw * D;
    float ss = 0.f;
    for (int d = lane; d < D; d += 32) { float v = xr[d]; ss += v * v; }
    for (int off = 16; off > 0; off >>= 1) ss += __shfl_xor(ss, off, 32);
    float scale = rsqrtf(ss / (float)D + 1e-5f);
    bf16_t* orow = out + (long)gw * D;
    for (int d = lane; d < D; d += 32) orow[d] = (bf16_t)(xr[d] * scale * w[d]);
}

// Causal depthwise conv (K=4) over the x-half of xz, + bias, + silu.
__global__ void k_conv_silu(const float* __restrict__ xz, const float* __restrict__ cw,
                            const float* __restrict__ cb, float* __restrict__ xc,
                            bf16_t* __restrict__ xcb) {
    int idx = blockIdx.x * blockDim.x + threadIdx.x;
    if (idx >= ROWS * EE) return;
    int e = idx % EE;
    int row = idx / EE;         // b*L + l
    int l = row % LL;
    int b = row / LL;
    float acc = cb[e];
    for (int j = 0; j < KK; ++j) {
        int li = l - (KK - 1) + j;
        if (li >= 0) acc += cw[e * KK + j] * xz[(long)(b * LL + li) * (2 * EE) + e];
    }
    float s = silu_f(acc);
    xc[idx] = s;
    xcb[idx] = (bf16_t)s;
}

__global__ void k_softplus_bias(float* __restrict__ dt, const float* __restrict__ dtb) {
    int idx = blockIdx.x * blockDim.x + threadIdx.x;
    if (idx >= ROWS * EE) return;
    int e = idx % EE;
    dt[idx] = softplus_f(dt[idx] + dtb[e]);
}

// ---------------------------------------------------------------------------
// Selective scan: one lane per (b, e) channel. State h[N=16] in registers.
// Per step each lane loads one of the 32 shared B/C values; broadcast via
// wave32 shuffles (no barriers in the 512-step loop). Fuses y*silu(z) -> bf16.
// ---------------------------------------------------------------------------
__global__ __launch_bounds__(256)
void k_scan(const float* __restrict__ xconv, const float* __restrict__ dt,
            const float* __restrict__ proj, const float* __restrict__ xz,
            const float* __restrict__ A_log, const float* __restrict__ Dp,
            bf16_t* __restrict__ ymul) {
    int gtid = blockIdx.x * blockDim.x + threadIdx.x;
    int gw = gtid >> 5;
    int lane = gtid & 31;
    const int wpb = EE / 32;
    int b = gw / wpb;
    if (b >= BB) return;
    int e = (gw % wpb) * 32 + lane;

    float A[NN], h[NN];
    for (int n = 0; n < NN; ++n) {
        A[n] = -__expf(A_log[e * NN + n]);
        h[n] = 0.f;
    }
    float d = Dp[e];

    for (int t = 0; t < LL; ++t) {
        long row = (long)b * LL + t;
        float x   = xconv[row * EE + e];
        float dtv = dt[row * EE + e];
        float z   = xz[row * (2 * EE) + EE + e];
        float bc  = proj[row * PROJW + RR + lane];  // lanes 0..15: B, 16..31: C
        float dx = dtv * x;
        float y = 0.f;
        for (int n = 0; n < NN; ++n) {
            float Bn = __shfl(bc, n, 32);
            float Cn = __shfl(bc, 16 + n, 32);
            float dA = __expf(dtv * A[n]);
            h[n] = dA * h[n] + dx * Bn;
            y += h[n] * Cn;
        }
        y += x * d;
        ymul[row * EE + e] = (bf16_t)(y * silu_f(z));
    }
}

// ---------------------------------------------------------------------------
// WMMA bf16 GEMM: C[M,N] = A[M,K] @ B[K,N] (+ residual), f32 accumulate.
// B is supplied PRE-TRANSPOSED: Bt[N,K] row-major (ldb = K), so both A and B
// tile staging are coalesced 16B-per-lane async global->LDS transfers.
// Double-buffered: tile kb+32 is prefetched (async) while WMMAs run on kb.
// Block = 128 threads (4 waves), tile 64(M) x 64(N), K stepped by 32.
// ---------------------------------------------------------------------------
#define TM 64
#define TN 64
#define TK 32
#define LDSP 40   // padded row stride (bf16 elems); 80B rows keep 16B alignment

__global__ __launch_bounds__(128)
void k_gemm_bf16(const bf16_t* __restrict__ A, int lda,
                 const bf16_t* __restrict__ Bt, int ldb,
                 float* __restrict__ C, int ldc,
                 const float* __restrict__ residual,
                 int M, int N, int Kdim) {
    __shared__ __align__(16) bf16_t As[2][TM * LDSP];
    __shared__ __align__(16) bf16_t Bs[2][TN * LDSP];

    const int tid = threadIdx.x;
    const int lane = tid & 31;
    const int wave = tid >> 5;              // 0..3 -> 16-row stripe of the tile
    const int m0 = blockIdx.x * TM;
    const int n0 = blockIdx.y * TN;

    const int row2 = tid >> 1;              // 0..63: staging row
    const int half = tid & 1;               // which 16-elem half of the 32-col row

    v8f acc[4];
    for (int s = 0; s < 4; ++s)
        for (int j = 0; j < 8; ++j) acc[s][j] = 0.f;

    const int mfrag = lane & 15;            // M (A) / N (B) index of this lane
    const int khalf = lane >> 4;            // which K-half this lane holds

    const bool mn_full = (m0 + TM <= M) && (n0 + TN <= N);

    // Stage one 64x32 A-tile + 64x32 B-tile into buffer `buf` for K-block kb.
    auto stage = [&](int kb, int buf) {
        bf16_t* asl = &As[buf][row2 * LDSP + half * 16];
        bf16_t* bsl = &Bs[buf][row2 * LDSP + half * 16];
        const bf16_t* ag = A  + (long)(m0 + row2) * lda + kb + half * 16;
        const bf16_t* bg = Bt + (long)(n0 + row2) * ldb + kb + half * 16;
        if (mn_full && (kb + TK <= Kdim)) {
#if USE_ASYNC_LDS
            __builtin_amdgcn_global_load_async_to_lds_b128((as1_v4i_p)ag, (as3_v4i_p)asl, 0, 0);
            __builtin_amdgcn_global_load_async_to_lds_b128((as1_v4i_p)bg, (as3_v4i_p)bsl, 0, 0);
#else
            *(uint4*)asl = *(const uint4*)ag;
            *(uint4*)bsl = *(const uint4*)bg;
#endif
        } else {
            // Edge tiles (N=80 / K=48 cases): guarded scalar staging, zero-fill.
            for (int j = 0; j < 16; ++j) {
                int k = kb + half * 16 + j;
                bf16_t av = (bf16_t)0.f, bv = (bf16_t)0.f;
                if ((m0 + row2) < M && k < Kdim) av = A[(long)(m0 + row2) * lda + k];
                if ((n0 + row2) < N && k < Kdim) bv = Bt[(long)(n0 + row2) * ldb + k];
                asl[j] = av;
                bsl[j] = bv;
            }
        }
    };

    // Prologue: stage first tile, wait, barrier.
    stage(0, 0);
#if USE_ASYNC_LDS
    __builtin_amdgcn_s_wait_asynccnt(0);
#endif
    __syncthreads();

    int buf = 0;
    for (int kb = 0; kb < Kdim; kb += TK) {
        const bool has_next = (kb + TK) < Kdim;
        // Prefetch next tile into the alternate buffer (overlaps the WMMAs).
        if (has_next) stage(kb + TK, buf ^ 1);

        // A fragment: elems 0..7 = K kh*8+0..7 ; elems 8..15 = K 16+kh*8+0..7
        v16bf a;
        {
            const bf16_t* ap = &As[buf][(wave * 16 + mfrag) * LDSP + khalf * 8];
            for (int j = 0; j < 8; ++j) { a[j] = ap[j]; a[8 + j] = ap[16 + j]; }
        }
        // All four B fragments first, then 4 back-to-back WMMAs.
        v16bf bvf[4];
        for (int s = 0; s < 4; ++s) {
            const bf16_t* bp = &Bs[buf][(s * 16 + mfrag) * LDSP + khalf * 8];
            for (int j = 0; j < 8; ++j) { bvf[s][j] = bp[j]; bvf[s][8 + j] = bp[16 + j]; }
        }
        for (int s = 0; s < 4; ++s) {
            acc[s] = __builtin_amdgcn_wmma_f32_16x16x32_bf16(
                false, a, false, bvf[s], (short)0, acc[s], false, false);
        }

#if USE_ASYNC_LDS
        if (has_next) __builtin_amdgcn_s_wait_asynccnt(0);
#endif
        __syncthreads();
        buf ^= 1;
    }

    // Store per C layout: lane n = lane&15; rows (lane>=16 ? 8:0)+r.
    const int nfrag = lane & 15;
    const int mbase = (lane >> 4) * 8;
    for (int s = 0; s < 4; ++s) {
        int gn = n0 + s * 16 + nfrag;
        if (gn >= N) continue;
        for (int r = 0; r < 8; ++r) {
            int gm = m0 + wave * 16 + mbase + r;
            if (gm >= M) continue;
            float v = acc[s][r];
            if (residual) v += residual[(long)gm * ldc + gn];
            C[(long)gm * ldc + gn] = v;
        }
    }
}

// ---------------------------------------------------------------------------
// Final norm + MLP head (tiny: 4 rows).
// ---------------------------------------------------------------------------
__global__ void k_cls_rmsnorm(const float* __restrict__ h, const float* __restrict__ w,
                              float* __restrict__ cls) {
    int wv = threadIdx.x >> 5;
    int lane = threadIdx.x & 31;
    if (wv >= BB) return;
    const float* xr = h + ((long)wv * LL + (LL - 1)) * DMODEL;
    float ss = 0.f;
    for (int d = lane; d < DMODEL; d += 32) { float v = xr[d]; ss += v * v; }
    for (int off = 16; off > 0; off >>= 1) ss += __shfl_xor(ss, off, 32);
    float scale = rsqrtf(ss / (float)DMODEL + 1e-5f);
    for (int d = lane; d < DMODEL; d += 32) cls[wv * DMODEL + d] = xr[d] * scale * w[d];
}

__global__ void k_head1(const float* __restrict__ cls, const float* __restrict__ tw,
                        const float* __restrict__ tb, float* __restrict__ hid) {
    int idx = blockIdx.x * blockDim.x + threadIdx.x;
    if (idx >= BB * HH) return;
    int b = idx / HH, j = idx % HH;
    float acc = tb[j];
    for (int d = 0; d < DMODEL; ++d) acc += cls[b * DMODEL + d] * tw[(long)d * HH + j];
    hid[idx] = fmaxf(acc, 0.f);
}

__global__ void k_head2(const float* __restrict__ hid, const float* __restrict__ bw,
                        const float* __restrict__ bb, float* __restrict__ out) {
    int idx = blockIdx.x * blockDim.x + threadIdx.x;
    if (idx >= BB * TT) return;
    int b = idx / TT, t = idx % TT;
    float acc = bb[t];
    for (int j = 0; j < HH; ++j) acc += hid[b * HH + j] * bw[(long)j * TT + t];
    out[idx] = acc;
}

// ---------------------------------------------------------------------------
// Host orchestration
// ---------------------------------------------------------------------------
extern "C" void kernel_launch(void* const* d_in, const int* in_sizes, int n_in,
                              void* d_out, int out_size, void* d_ws, size_t ws_size,
                              hipStream_t stream) {
    const int*   input_ids = (const int*)d_in[0];
    const float* embed_w   = (const float*)d_in[1];
    const float* in_proj_w = (const float*)d_in[2];
    const float* conv_w    = (const float*)d_in[3];
    const float* conv_b    = (const float*)d_in[4];
    const float* x_proj_w  = (const float*)d_in[5];
    const float* dt_w      = (const float*)d_in[6];
    const float* dt_b      = (const float*)d_in[7];
    const float* A_log     = (const float*)d_in[8];
    const float* D_p       = (const float*)d_in[9];
    const float* out_w     = (const float*)d_in[10];
    const float* norm_w    = (const float*)d_in[11];
    const float* norm_f_w  = (const float*)d_in[12];
    const float* top_w     = (const float*)d_in[13];
    const float* top_b     = (const float*)d_in[14];
    const float* bot_w     = (const float*)d_in[15];
    const float* bot_b     = (const float*)d_in[16];
    float* outp = (float*)d_out;

    char* p = (char*)d_ws;
    auto alloc = [&](size_t bytes) -> void* {
        void* r = (void*)p;
        p += (bytes + 255) & ~(size_t)255;
        return r;
    };
    float*  h      = (float*) alloc(sizeof(float)  * (size_t)ROWS * DMODEL);
    float*  xz     = (float*) alloc(sizeof(float)  * (size_t)ROWS * 2 * EE);
    float*  xconv  = (float*) alloc(sizeof(float)  * (size_t)ROWS * EE);
    float*  dt     = (float*) alloc(sizeof(float)  * (size_t)ROWS * EE);
    float*  proj   = (float*) alloc(sizeof(float)  * (size_t)ROWS * PROJW);
    float*  cls    = (float*) alloc(sizeof(float)  * BB * DMODEL);
    float*  hid    = (float*) alloc(sizeof(float)  * BB * HH);
    bf16_t* xib    = (bf16_t*)alloc(sizeof(bf16_t) * (size_t)ROWS * DMODEL);
    bf16_t* xcb    = (bf16_t*)alloc(sizeof(bf16_t) * (size_t)ROWS * EE);
    bf16_t* projb  = (bf16_t*)alloc(sizeof(bf16_t) * (size_t)ROWS * PROJW);
    bf16_t* ymulb  = (bf16_t*)alloc(sizeof(bf16_t) * (size_t)ROWS * EE);
    bf16_t* w_inT  = (bf16_t*)alloc(sizeof(bf16_t) * (size_t)DMODEL * 2 * EE);  // [2E x D]
    bf16_t* w_xpT  = (bf16_t*)alloc(sizeof(bf16_t) * (size_t)EE * PROJW);       // [80 x E]
    bf16_t* w_dtT  = (bf16_t*)alloc(sizeof(bf16_t) * (size_t)RR * EE);          // [E x R]
    bf16_t* w_outT = (bf16_t*)alloc(sizeof(bf16_t) * (size_t)EE * DMODEL);      // [D x E]

    auto cvt = [&](const float* src, bf16_t* dst, int n) {
        k_f32_to_bf16<<<(n + 255) / 256, 256, 0, stream>>>(src, dst, n);
    };
    auto cvtT = [&](const float* src, bf16_t* dst, int Kd, int Nd) {
        int n = Kd * Nd;
        k_f32_to_bf16_T<<<(n + 255) / 256, 256, 0, stream>>>(src, dst, Kd, Nd);
    };
    auto gemm = [&](const bf16_t* A, int lda, const bf16_t* BtM,
                    float* C, int ldc, const float* resid, int M, int N, int Kd) {
        dim3 g((M + TM - 1) / TM, (N + TN - 1) / TN);
        k_gemm_bf16<<<g, 128, 0, stream>>>(A, lda, BtM, Kd, C, ldc, resid, M, N, Kd);
    };

    // h = embed[input_ids]
    k_embed<<<(ROWS * DMODEL + 255) / 256, 256, 0, stream>>>(input_ids, embed_w, h);

    for (int i = 0; i < NLAYER; ++i) {
        // xi = rmsnorm(h, norm_w[i]) -> bf16
        k_rmsnorm_bf16<<<(ROWS * 32 + 255) / 256, 256, 0, stream>>>(
            h, norm_w + (size_t)i * DMODEL, xib, ROWS, DMODEL);

        // xz = xi @ in_proj_w[i]   (2048 x 768 x 3072)
        cvtT(in_proj_w + (size_t)i * DMODEL * 2 * EE, w_inT, DMODEL, 2 * EE);
        gemm(xib, DMODEL, w_inT, xz, 2 * EE, nullptr, ROWS, 2 * EE, DMODEL);

        // x = silu(depthwise_conv(x) + b)
        k_conv_silu<<<(ROWS * EE + 255) / 256, 256, 0, stream>>>(
            xz, conv_w + (size_t)i * EE * KK, conv_b + (size_t)i * EE, xconv, xcb);

        // proj = x @ x_proj_w[i]   (2048 x 1536 x 80)
        cvtT(x_proj_w + (size_t)i * EE * PROJW, w_xpT, EE, PROJW);
        gemm(xcb, EE, w_xpT, proj, PROJW, nullptr, ROWS, PROJW, EE);

        // dt = softplus(proj[:, :48] @ dt_w[i] + dt_b[i])   (2048 x 48 x 1536)
        cvt(proj, projb, ROWS * PROJW);
        cvtT(dt_w + (size_t)i * RR * EE, w_dtT, RR, EE);
        gemm(projb, PROJW, w_dtT, dt, EE, nullptr, ROWS, EE, RR);
        k_softplus_bias<<<(ROWS * EE + 255) / 256, 256, 0, stream>>>(
            dt, dt_b + (size_t)i * EE);

        // selective scan (fused with y * silu(z) -> bf16)
        k_scan<<<(BB * EE + 255) / 256, 256, 0, stream>>>(
            xconv, dt, proj, xz, A_log + (size_t)i * EE * NN, D_p + (size_t)i * EE, ymulb);

        // h = h + ymul @ out_w[i]  (2048 x 1536 x 768, fused residual)
        cvtT(out_w + (size_t)i * EE * DMODEL, w_outT, EE, DMODEL);
        gemm(ymulb, EE, w_outT, h, DMODEL, h, ROWS, DMODEL, EE);
    }

    // final rmsnorm on last-token rows, then 2-layer head
    k_cls_rmsnorm<<<1, 128, 0, stream>>>(h, norm_f_w, cls);
    k_head1<<<(BB * HH + 255) / 256, 256, 0, stream>>>(cls, top_w, top_b, hid);
    k_head2<<<1, 128, 0, stream>>>(hid, bot_w, bot_b, outp);
}
